// SimpleSSMDynamics_41042707481104
// MI455X (gfx1250) — matrix operand order repaired and checked by reading, not compiled, not run on hardware
//
#include <hip/hip_runtime.h>
#include <hip/hip_bf16.h>

#define BDIM   16
#define TLEN   2048
#define OBS    64
#define ACTD   16
#define HID    512
#define KIN    (OBS + ACTD)     // 80
#define KPAD   96               // pad K to multiple of 32
#define BT     (BDIM * TLEN)    // 32768 rows
#define NCHUNK 16
#define CLEN   (TLEN / NCHUNK)  // 128

typedef __attribute__((ext_vector_type(16))) __bf16 v16bf;
typedef __attribute__((ext_vector_type(8)))  __bf16 v8bf;
typedef __attribute__((ext_vector_type(8)))  float  v8f;

// ---------------------------------------------------------------------------
// Prep kernels: pack x=[obs|act|0pad] to bf16, convert weights, tanh(a_raw)
// ---------------------------------------------------------------------------
__global__ __launch_bounds__(256) void pack_x_kernel(
    const float* __restrict__ obs, const float* __restrict__ act,
    __bf16* __restrict__ x96)
{
    const int i = blockIdx.x * blockDim.x + threadIdx.x;
    if (i >= BT * KPAD) return;
    const int row = i / KPAD;
    const int c   = i - row * KPAD;
    float v = 0.0f;
    if (c < OBS)      v = obs[(size_t)row * OBS + c];
    else if (c < KIN) v = act[(size_t)row * ACTD + (c - OBS)];
    x96[i] = (__bf16)v;
}

__global__ __launch_bounds__(256) void cvt_we_kernel(
    const float* __restrict__ We, __bf16* __restrict__ We96)
{
    const int i = blockIdx.x * blockDim.x + threadIdx.x;
    if (i >= HID * KPAD) return;
    const int r = i / KPAD;
    const int c = i - r * KPAD;
    We96[i] = (__bf16)((c < KIN) ? We[(size_t)r * KIN + c] : 0.0f);
}

__global__ __launch_bounds__(256) void cvt_bf16_kernel(
    const float* __restrict__ src, __bf16* __restrict__ dst, int n)
{
    const int i = blockIdx.x * blockDim.x + threadIdx.x;
    if (i < n) dst[i] = (__bf16)src[i];
}

__global__ __launch_bounds__(256) void tanh_kernel(
    const float* __restrict__ a_raw, float* __restrict__ a, int n)
{
    const int i = blockIdx.x * blockDim.x + threadIdx.x;
    if (i < n) a[i] = tanhf(a_raw[i]);
}

// ---------------------------------------------------------------------------
// WMMA GEMM:  Out[M,N] = act( A[M,K] * W[N,K]^T + bias[N] )
// One wave -> 16x64 output strip (4 accumulators, shared A fragment).
//
// Fragment layouts used here (contiguous k order, same for A and B, wave32):
//  A 16x32 bf16 : lane l (m=l&15, hi=l>>4) holds K = 16*hi..16*hi+15
//                 -> two contiguous b128 loads
//  B 32x16 bf16 : lane l (n=l&15, hi=l>>4) holds K = 16*hi..16*hi+15
//                 -> 16 contiguous bf16 from W row n  (W is [N,K] row-major)
//  C/D 16x16 f32: lane l covers col l&15, rows r + 8*(l>>4), r=0..7
// ---------------------------------------------------------------------------
template<bool RELU, bool OUT_BF16, bool HAS_BIAS>
__global__ __launch_bounds__(256) void gemm_bf16_wmma(
    const __bf16* __restrict__ A, int lda,
    const __bf16* __restrict__ W, int ldw,
    const float*  __restrict__ bias,
    void* __restrict__ Out, int ldo,
    int M, int N, int K)
{
    const int lane   = threadIdx.x & 31;
    const int waveId = (int)((blockIdx.x * blockDim.x + threadIdx.x) >> 5);
    const int nT64   = N >> 6;
    const int mTile  = waveId / nT64;
    const int n64    = waveId - mTile * nT64;
    if ((mTile << 4) >= M) return;   // wave-uniform: EXEC stays all-ones for WMMA

    const int l15     = lane & 15;
    const int hi      = lane >> 4;
    const int colBase = (n64 << 6) + l15;

    const __bf16* arow = A + (size_t)((mTile << 4) + l15) * lda;
    const __bf16* w0   = W + (size_t)(colBase +  0) * ldw;
    const __bf16* w1   = W + (size_t)(colBase + 16) * ldw;
    const __bf16* w2   = W + (size_t)(colBase + 32) * ldw;
    const __bf16* w3   = W + (size_t)(colBase + 48) * ldw;

    v8f acc0 = {}, acc1 = {}, acc2 = {}, acc3 = {};

    for (int k0 = 0; k0 < K; k0 += 32) {
        const v8bf alo = *(const v8bf*)(arow + k0 + 16 * hi);
        const v8bf ahi = *(const v8bf*)(arow + k0 + 16 * hi + 8);
        const v16bf af = __builtin_shufflevector(alo, ahi,
            0,1,2,3,4,5,6,7,8,9,10,11,12,13,14,15);

        const int kb = k0 + 16 * hi;
        const v16bf b0 = *(const v16bf*)(w0 + kb);
        const v16bf b1 = *(const v16bf*)(w1 + kb);
        const v16bf b2 = *(const v16bf*)(w2 + kb);
        const v16bf b3 = *(const v16bf*)(w3 + kb);

        acc0 = __builtin_amdgcn_wmma_f32_16x16x32_bf16(false, af, false, b0, (short)0, acc0, false, false);
        acc1 = __builtin_amdgcn_wmma_f32_16x16x32_bf16(false, af, false, b1, (short)0, acc1, false, false);
        acc2 = __builtin_amdgcn_wmma_f32_16x16x32_bf16(false, af, false, b2, (short)0, acc2, false, false);
        acc3 = __builtin_amdgcn_wmma_f32_16x16x32_bf16(false, af, false, b3, (short)0, acc3, false, false);
    }

    const int rowBase = (mTile << 4) + 8 * hi;
    v8f accs[4] = {acc0, acc1, acc2, acc3};
    #pragma unroll
    for (int s = 0; s < 4; ++s) {
        const int col = colBase + 16 * s;
        const float bv = HAS_BIAS ? bias[col] : 0.0f;
        #pragma unroll
        for (int r = 0; r < 8; ++r) {
            float val = accs[s][r] + bv;
            if (RELU) val = fmaxf(val, 0.0f);
            const size_t idx = (size_t)(rowBase + r) * ldo + col;
            if (OUT_BF16) ((__bf16*)Out)[idx] = (__bf16)val;
            else          ((float*)Out)[idx]  = val;
        }
    }
}

// ---------------------------------------------------------------------------
// Chunked diagonal scan: s_t = a*s_{t-1} + v_t over T, per (b,h).
// Pass1: per-chunk local reduction -> carry.  Pass2: sequential scan of
// NCHUNK carries per (b,h).  Pass3: recompute chunk scans with true init
// state, emit s as bf16 for the output GEMM.
// ---------------------------------------------------------------------------
__global__ __launch_bounds__(256) void scan_pass1(
    const float* __restrict__ v, const float* __restrict__ a,
    float* __restrict__ carry)
{
    const int tid = blockIdx.x * blockDim.x + threadIdx.x; // < B*NCHUNK*H
    const int h   = tid & (HID - 1);
    const int bc  = tid / HID;              // b*NCHUNK + c
    const int c   = bc & (NCHUNK - 1);
    const int b   = bc / NCHUNK;
    const float ah = a[h];
    const float* vp = v + (size_t)(b * TLEN + c * CLEN) * HID + h;
    float s = 0.0f;
    #pragma unroll 4
    for (int i = 0; i < CLEN; ++i) s = fmaf(ah, s, vp[(size_t)i * HID]);
    carry[(size_t)bc * HID + h] = s;
}

__global__ __launch_bounds__(256) void scan_pass2(
    const float* __restrict__ carry, const float* __restrict__ a,
    float* __restrict__ cin)
{
    const int tid = blockIdx.x * blockDim.x + threadIdx.x; // < B*H
    const int h = tid & (HID - 1);
    const int b = tid / HID;
    const float ah = a[h];
    float d = ah;
    #pragma unroll
    for (int j = 0; j < 7; ++j) d *= d;     // ah^128 == ah^CLEN
    float st = 0.0f;
    for (int c = 0; c < NCHUNK; ++c) {
        const size_t idx = (size_t)(b * NCHUNK + c) * HID + h;
        cin[idx] = st;
        st = fmaf(d, st, carry[idx]);
    }
}

__global__ __launch_bounds__(256) void scan_pass3(
    const float* __restrict__ v, const float* __restrict__ a,
    const float* __restrict__ cin, __bf16* __restrict__ sb)
{
    const int tid = blockIdx.x * blockDim.x + threadIdx.x; // < B*NCHUNK*H
    const int h   = tid & (HID - 1);
    const int bc  = tid / HID;
    const int c   = bc & (NCHUNK - 1);
    const int b   = bc / NCHUNK;
    const float ah = a[h];
    float s = cin[(size_t)bc * HID + h];
    const size_t base = (size_t)(b * TLEN + c * CLEN) * HID + h;
    #pragma unroll 4
    for (int i = 0; i < CLEN; ++i) {
        s = fmaf(ah, s, v[base + (size_t)i * HID]);
        sb[base + (size_t)i * HID] = (__bf16)s;
    }
}

// ---------------------------------------------------------------------------
// Launch
// ---------------------------------------------------------------------------
extern "C" void kernel_launch(void* const* d_in, const int* in_sizes, int n_in,
                              void* d_out, int out_size, void* d_ws, size_t ws_size,
                              hipStream_t stream)
{
    const float* obs   = (const float*)d_in[0];
    const float* act   = (const float*)d_in[1];
    const float* W_e   = (const float*)d_in[2];
    const float* b_e   = (const float*)d_in[3];
    const float* a_raw = (const float*)d_in[4];
    const float* W_B   = (const float*)d_in[5];
    const float* W_out = (const float*)d_in[6];
    const float* b_out = (const float*)d_in[7];
    float* out = (float*)d_out;

    char* ws = (char*)d_ws;
    size_t off = 0;
    auto take = [&](size_t bytes) -> char* {
        char* p = ws + off;
        off += (bytes + 255) & ~(size_t)255;
        return p;
    };
    __bf16* x96   = (__bf16*)take((size_t)BT  * KPAD * 2);  //  6.0 MB
    __bf16* We96  = (__bf16*)take((size_t)HID * KPAD * 2);  //  0.1 MB
    __bf16* Wb16  = (__bf16*)take((size_t)HID * HID  * 2);  //  0.5 MB
    __bf16* Wo16  = (__bf16*)take((size_t)OBS * HID  * 2);  //  0.06 MB
    float*  aT    = (float*) take((size_t)HID * 4);
    __bf16* u     = (__bf16*)take((size_t)BT * HID * 2);    // 32 MB
    float*  vbuf  = (float*) take((size_t)BT * HID * 4);    // 64 MB
    __bf16* sbuf  = (__bf16*)take((size_t)BT * HID * 2);    // 32 MB
    float*  carry = (float*) take((size_t)BDIM * NCHUNK * HID * 4);
    float*  cin   = (float*) take((size_t)BDIM * NCHUNK * HID * 4);
    (void)in_sizes; (void)n_in; (void)out_size; (void)ws_size;

    // --- prep ---
    pack_x_kernel  <<<(BT * KPAD + 255) / 256, 256, 0, stream>>>(obs, act, x96);
    cvt_we_kernel  <<<(HID * KPAD + 255) / 256, 256, 0, stream>>>(W_e, We96);
    cvt_bf16_kernel<<<(HID * HID + 255) / 256, 256, 0, stream>>>(W_B, Wb16, HID * HID);
    cvt_bf16_kernel<<<(OBS * HID + 255) / 256, 256, 0, stream>>>(W_out, Wo16, OBS * HID);
    tanh_kernel    <<<(HID + 255) / 256, 256, 0, stream>>>(a_raw, aT, HID);

    // --- GEMM1: u = relu(x @ W_e^T + b_e), bf16 out ---
    {
        const int waves = (BT / 16) * (HID / 64);  // 16384
        gemm_bf16_wmma<true, true, true><<<waves / 8, 256, 0, stream>>>(
            x96, KPAD, We96, KPAD, b_e, (void*)u, HID, BT, HID, KPAD);
    }
    // --- GEMM2: v = u @ W_B^T, f32 out (feeds the scan) ---
    {
        const int waves = (BT / 16) * (HID / 64);
        gemm_bf16_wmma<false, false, false><<<waves / 8, 256, 0, stream>>>(
            u, HID, Wb16, HID, nullptr, (void*)vbuf, HID, BT, HID, HID);
    }
    // --- chunked scan ---
    scan_pass1<<<(BDIM * NCHUNK * HID) / 256, 256, 0, stream>>>(vbuf, aT, carry);
    scan_pass2<<<(BDIM * HID) / 256, 256, 0, stream>>>(carry, aT, cin);
    scan_pass3<<<(BDIM * NCHUNK * HID) / 256, 256, 0, stream>>>(vbuf, aT, cin, sbuf);

    // --- GEMM3: preds = s @ W_out^T + b_out, f32 -> d_out ---
    {
        const int waves = (BT / 16) * (OBS / 64);  // 2048
        gemm_bf16_wmma<false, false, true><<<waves / 8, 256, 0, stream>>>(
            sbuf, HID, Wo16, HID, b_out, (void*)out, OBS, BT, OBS, HID);
    }
}
